// NERHeadWithCRF_19670950216415
// MI455X (gfx1250) — compile-verified, matrix operand order checked by reading, not run on hardware
//
#include <hip/hip_runtime.h>
#include <hip/hip_bf16.h>

typedef unsigned short u16;
typedef __attribute__((ext_vector_type(16))) __bf16 v16bf;
typedef __attribute__((ext_vector_type(16))) unsigned short v16u;
typedef __attribute__((ext_vector_type(8)))  float v8f;

#define B_  16
#define S_  1024
#define D_  4096
#define W_  512
#define H_  256
#define P_  512      // projection output
#define G4_ 1024     // 4*H
#define C_  7
#define M_  (B_ * W_) // 8192 flattened (batch, word) rows

// ---------- helpers ----------
__device__ inline u16 f2bf(float f) {            // native RTNE convert
    __bf16 h = (__bf16)f;
    return __builtin_bit_cast(u16, h);
}
__device__ inline float sigm(float x) { return 1.0f / (1.0f + expf(-x)); }
__device__ inline float gelu(float x) { return 0.5f * x * (1.0f + erff(x * 0.70710678118654752f)); }

union FU { v16u v; u16 a[16]; };
union FB { v16bf v; __bf16 a[16]; };

// raw 16x32 bf16 fragment data: two 16B chunks (k..k+7 and k+16..k+23)
struct BRaw { uint4 lo, hi; };
__device__ inline BRaw load_braw(const u16* p) {
    const uint4* q = reinterpret_cast<const uint4*>(p);
    BRaw r; r.lo = q[0]; r.hi = q[2];
    return r;
}
__device__ inline v16bf braw_frag(const BRaw& r) {
    FU u;
    u.a[0]=(u16)r.lo.x;  u.a[1]=(u16)(r.lo.x>>16);
    u.a[2]=(u16)r.lo.y;  u.a[3]=(u16)(r.lo.y>>16);
    u.a[4]=(u16)r.lo.z;  u.a[5]=(u16)(r.lo.z>>16);
    u.a[6]=(u16)r.lo.w;  u.a[7]=(u16)(r.lo.w>>16);
    u.a[8]=(u16)r.hi.x;  u.a[9]=(u16)(r.hi.x>>16);
    u.a[10]=(u16)r.hi.y; u.a[11]=(u16)(r.hi.y>>16);
    u.a[12]=(u16)r.hi.z; u.a[13]=(u16)(r.hi.z>>16);
    u.a[14]=(u16)r.hi.w; u.a[15]=(u16)(r.hi.w>>16);
    return __builtin_bit_cast(v16bf, u.v);
}
__device__ inline v16bf frag_bf16(const u16* p) { return braw_frag(load_braw(p)); }

// raw f32 A-fragment data (32 B at +0, 32 B at +64 bytes)
struct ARaw { float4 x, y, z, w; };
__device__ inline ARaw load_araw(const float* p) {
    const float4* q = reinterpret_cast<const float4*>(p);
    ARaw r; r.x = q[0]; r.y = q[1]; r.z = q[4]; r.w = q[5];
    return r;
}
__device__ inline v16bf araw_conv(const ARaw& r, float s) {
    FB u;
    u.a[0]=(__bf16)(r.x.x*s); u.a[1]=(__bf16)(r.x.y*s);
    u.a[2]=(__bf16)(r.x.z*s); u.a[3]=(__bf16)(r.x.w*s);
    u.a[4]=(__bf16)(r.y.x*s); u.a[5]=(__bf16)(r.y.y*s);
    u.a[6]=(__bf16)(r.y.z*s); u.a[7]=(__bf16)(r.y.w*s);
    u.a[8]=(__bf16)(r.z.x*s); u.a[9]=(__bf16)(r.z.y*s);
    u.a[10]=(__bf16)(r.z.z*s); u.a[11]=(__bf16)(r.z.w*s);
    u.a[12]=(__bf16)(r.w.x*s); u.a[13]=(__bf16)(r.w.y*s);
    u.a[14]=(__bf16)(r.w.z*s); u.a[15]=(__bf16)(r.w.w*s);
    return u.v;
}

// load all 8 N-subtile B fragments of one k-step (constant nt*16*K offsets)
template<int K>
__device__ inline void load_b8(BRaw (&br)[8], const u16* __restrict__ bb_k) {
#pragma unroll
    for (int nt = 0; nt < 8; ++nt) br[nt] = load_braw(bb_k + (size_t)nt * 16 * K);
}
// issue the 8 WMMAs of one k-step
__device__ inline void wmma8(const BRaw (&br)[8], v16bf af, v8f (&acc)[8]) {
#pragma unroll
    for (int nt = 0; nt < 8; ++nt) {
        acc[nt] = __builtin_amdgcn_wmma_f32_16x16x32_bf16(
            false, af, false, braw_frag(br[nt]), (short)0, acc[nt], false, false);
    }
}

// ---------- prep kernels ----------
__global__ void k_init_first(int* first) {
    int i = blockIdx.x * blockDim.x + threadIdx.x;
    if (i < B_ * W_) first[i] = S_;
}
__global__ void k_scan_first(const int* wi, int* first) {
    int i = blockIdx.x * blockDim.x + threadIdx.x;
    if (i >= B_ * S_) return;
    int b = i >> 10, s = i & (S_ - 1);
    int w = wi[i];
    int pos = (w >= 0) ? s : S_;
    int c = min(max(w, 0), W_ - 1);
    atomicMin(&first[(b << 9) + c], pos);
}
__global__ void k_mask(const int* first, int* mask) {
    int i = blockIdx.x * blockDim.x + threadIdx.x;
    if (i < B_ * W_) mask[i] = (first[i] < S_) ? 1 : 0;
}
__global__ void k_f2bf_arr(const float* in, u16* out, int n) {
    int i = blockIdx.x * blockDim.x + threadIdx.x;
    if (i < n) out[i] = f2bf(in[i]);
}

// ---------- projection GEMM: gather(features) @ proj_w^T + b, GELU, -> bf16 ----------
// block: 256 thr (8 waves: 2 M x 4 N). block tile 32(M) x 512(N). grid (N/512, M/32)
__global__ __launch_bounds__(256) void k_proj_gemm(
    const float* __restrict__ features, const int* __restrict__ first,
    const u16* __restrict__ Bw, const float* __restrict__ bias,
    u16* __restrict__ Cbf)
{
    const int lane = threadIdx.x & 31, wv = threadIdx.x >> 5;
    const int wm = wv & 1, wn = wv >> 1;
    const int rowBase = blockIdx.y * 32 + wm * 16;
    const int colBase = blockIdx.x * 512 + wn * 128;
    const int kb = (lane >> 4) << 3;       // 0 or 8
    const int cl = lane & 15;

    // per-lane gathered A row
    const int mrow = rowBase + cl;
    const int bb = mrow >> 9, w = mrow & (W_ - 1);
    const int f = first[(bb << 9) + w];
    const float msk = (f < S_) ? 1.0f : 0.0f;
    const int g = min(f, S_ - 1);
    const float* abase = features + ((size_t)bb * S_ + g) * D_ + kb;
    const u16* bbase = Bw + (size_t)(colBase + cl) * D_ + kb;

    v8f acc[8];
    const v8f vz = {0.f,0.f,0.f,0.f,0.f,0.f,0.f,0.f};
#pragma unroll
    for (int nt = 0; nt < 8; ++nt) acc[nt] = vz;

    // deep software pipeline: B ping-pong one k-step ahead, A two k-steps ahead
    BRaw b0[8], b1[8];
    load_b8<D_>(b0, bbase);                 // k = 0
    ARaw ar0 = load_araw(abase);            // k = 0
    ARaw ar1 = load_araw(abase + 32);       // k = 32
    for (int k0 = 0; k0 < D_; k0 += 64) {
        v16bf af0 = araw_conv(ar0, msk);
        load_b8<D_>(b1, bbase + k0 + 32);               // B for step k0+32
        if (k0 + 64 < D_) ar0 = load_araw(abase + k0 + 64);
        wmma8(b0, af0, acc);                            // waits on prev-step loads
        v16bf af1 = araw_conv(ar1, msk);
        if (k0 + 64 < D_) {
            load_b8<D_>(b0, bbase + k0 + 64);           // B for step k0+64
            ar1 = load_araw(abase + k0 + 96);
        }
        wmma8(b1, af1, acc);
    }

#pragma unroll
    for (int nt = 0; nt < 8; ++nt) {
        int col = colBase + nt * 16 + cl;
        float bv = bias[col];
#pragma unroll
        for (int r = 0; r < 8; ++r) {
            int row = rowBase + r + kb;
            float v = gelu(acc[nt][r] + bv);
            Cbf[(size_t)row * P_ + col] = f2bf(v);
        }
    }
}

// ---------- generic bf16 GEMM: C = A(MxK) @ Bw(NxK)^T + bias0 + bias1 -> f32 ----------
template<int K, int N>
__global__ __launch_bounds__(256) void k_gemm_bf16(
    const u16* __restrict__ A, const u16* __restrict__ Bw,
    const float* __restrict__ bias0, const float* __restrict__ bias1,
    float* __restrict__ Cf)
{
    const int lane = threadIdx.x & 31, wv = threadIdx.x >> 5;
    const int wm = wv & 1, wn = wv >> 1;
    const int rowBase = blockIdx.y * 32 + wm * 16;
    const int colBase = blockIdx.x * 512 + wn * 128;
    const int kb = (lane >> 4) << 3;
    const int cl = lane & 15;

    const u16* abase = A + (size_t)(rowBase + cl) * K + kb;
    const u16* bbase = Bw + (size_t)(colBase + cl) * K + kb;

    v8f acc[8];
    const v8f vz = {0.f,0.f,0.f,0.f,0.f,0.f,0.f,0.f};
#pragma unroll
    for (int nt = 0; nt < 8; ++nt) acc[nt] = vz;

    BRaw b0[8], b1[8];
    load_b8<K>(b0, bbase);                  // k = 0
    BRaw ar0 = load_braw(abase);            // k = 0
    BRaw ar1 = load_braw(abase + 32);       // k = 32
    for (int k0 = 0; k0 < K; k0 += 64) {
        v16bf af0 = braw_frag(ar0);
        load_b8<K>(b1, bbase + k0 + 32);
        if (k0 + 64 < K) ar0 = load_braw(abase + k0 + 64);
        wmma8(b0, af0, acc);
        v16bf af1 = braw_frag(ar1);
        if (k0 + 64 < K) {
            load_b8<K>(b0, bbase + k0 + 64);
            ar1 = load_braw(abase + k0 + 96);
        }
        wmma8(b1, af1, acc);
    }

#pragma unroll
    for (int nt = 0; nt < 8; ++nt) {
        int col = colBase + nt * 16 + cl;
        float bv = bias0[col] + bias1[col];
#pragma unroll
        for (int rr = 0; rr < 8; ++rr) {
            int row = rowBase + rr + kb;
            Cf[(size_t)row * N + col] = acc[nt][rr] + bv;
        }
    }
}

// ---------- persistent BiLSTM recurrent kernel: 2 blocks (fwd/bwd), 256 thr ----------
__global__ __launch_bounds__(256) void k_lstm(
    const float* __restrict__ xg_f, const float* __restrict__ xg_b,
    const u16* __restrict__ whh_f, const u16* __restrict__ whh_b,
    float* __restrict__ hf, float* __restrict__ hb)
{
    const int dir = blockIdx.x;
    const float* xg = dir ? xg_b : xg_f;
    const u16* whh = dir ? whh_b : whh_f;
    float* hout = dir ? hb : hf;

    __shared__ __align__(16) float g_s[B_ * G4_];   // 64 KB gate pre-activations
    __shared__ __align__(16) float c_s[B_ * H_];    // 16 KB cell state
    __shared__ __align__(16) u16   h_s[B_ * H_];    // 8 KB hidden (bf16, A-matrix)

    const int tid = threadIdx.x;
    const int lane = tid & 31, wv = tid >> 5;
    const int kb = (lane >> 4) << 3;
    const int cl = lane & 15;

    for (int i = tid; i < B_ * H_; i += 256) { c_s[i] = 0.0f; h_s[i] = 0; }
    __syncthreads();

    for (int step = 0; step < W_; ++step) {
        const int t = dir ? (W_ - 1 - step) : step;
        // g = h @ whh^T + xg[t] ; wave wv owns n in [wv*128, wv*128+128)
#pragma unroll
        for (int nt = 0; nt < 8; ++nt) {
            const int n0 = wv * 128 + nt * 16;
            const u16* bb = whh + (size_t)(n0 + cl) * H_ + kb;
            v8f acc = {0.f,0.f,0.f,0.f,0.f,0.f,0.f,0.f};
            BRaw br[2];
            br[0] = load_braw(bb);
#pragma unroll
            for (int kk = 0; kk < 8; ++kk) {       // K = 256 = 8 x 32
                if (kk < 7) br[(kk + 1) & 1] = load_braw(bb + (kk + 1) * 32);
                v16bf af = frag_bf16(&h_s[cl * H_ + kk * 32 + kb]);   // ds_read_b128
                acc = __builtin_amdgcn_wmma_f32_16x16x32_bf16(
                    false, af, false, braw_frag(br[kk & 1]), (short)0, acc, false, false);
            }
            const int col = n0 + cl;
#pragma unroll
            for (int r = 0; r < 8; ++r) {
                int bbat = r + kb;
                g_s[bbat * G4_ + col] =
                    acc[r] + xg[((size_t)bbat * W_ + t) * G4_ + col];
            }
        }
        __syncthreads();
        // gate nonlinearity + state update
        for (int idx = tid; idx < B_ * H_; idx += 256) {
            int bbat = idx >> 8, j = idx & (H_ - 1);
            float gi = g_s[bbat * G4_ + j];
            float gf = g_s[bbat * G4_ + H_ + j];
            float gg = g_s[bbat * G4_ + 2 * H_ + j];
            float go = g_s[bbat * G4_ + 3 * H_ + j];
            float c = sigm(gf) * c_s[idx] + sigm(gi) * tanhf(gg);
            float h = sigm(go) * tanhf(c);
            c_s[idx] = c;
            h_s[idx] = f2bf(h);
            hout[((size_t)bbat * W_ + t) * H_ + j] = h;
        }
        __syncthreads();
    }
}

// ---------- classifier: emissions = [hf,hb] @ cls_w^T + cls_b ----------
__global__ void k_cls(const float* __restrict__ hf, const float* __restrict__ hb,
                      const float* __restrict__ cw, const float* __restrict__ cb,
                      float* __restrict__ emis)
{
    int idx = blockIdx.x * blockDim.x + threadIdx.x;   // (b*W + t)
    if (idx >= M_) return;
    float acc[C_];
#pragma unroll
    for (int c = 0; c < C_; ++c) acc[c] = cb[c];
    const float* pf = hf + (size_t)idx * H_;
    const float* pb = hb + (size_t)idx * H_;
    for (int j = 0; j < H_; ++j) {
        float vf = pf[j], vb = pb[j];
#pragma unroll
        for (int c = 0; c < C_; ++c) {
            acc[c] += vf * cw[c * (2 * H_) + j] + vb * cw[c * (2 * H_) + H_ + j];
        }
    }
#pragma unroll
    for (int c = 0; c < C_; ++c) emis[(size_t)idx * C_ + c] = acc[c];
}

// ---------- Viterbi: one block per batch row ----------
__global__ void k_viterbi(const float* __restrict__ emis, const int* __restrict__ mask,
                          const float* __restrict__ st, const float* __restrict__ et,
                          const float* __restrict__ tr, int* __restrict__ bp,
                          float* __restrict__ out)
{
    const int b = blockIdx.x, tid = threadIdx.x;
    __shared__ float sc[C_], scn[C_];
    __shared__ int seq_end;
    __shared__ int tagbuf[W_];

    if (tid == 0) {
        int s = 0;
        for (int t = 0; t < W_; ++t) s += mask[b * W_ + t];
        seq_end = s - 1;
    }
    if (tid < C_) sc[tid] = st[tid] + emis[((size_t)b * W_) * C_ + tid];
    __syncthreads();

    for (int t = 1; t < W_; ++t) {
        if (tid < C_) {
            float best = -3.4e38f; int arg = 0;
            for (int cf = 0; cf < C_; ++cf) {
                float v = sc[cf] + tr[cf * C_ + tid];
                if (v > best) { best = v; arg = cf; }
            }
            best += emis[((size_t)b * W_ + t) * C_ + tid];
            bp[((size_t)b * W_ + t) * C_ + tid] = arg;
            scn[tid] = mask[b * W_ + t] ? best : sc[tid];
        }
        __syncthreads();
        if (tid < C_) sc[tid] = scn[tid];
        __syncthreads();
    }
    if (tid < C_) sc[tid] += et[tid];
    __syncthreads();

    if (tid == 0) {
        float best = -3.4e38f; int bt = 0;
        for (int c = 0; c < C_; ++c) if (sc[c] > best) { best = sc[c]; bt = c; }
        out[2 * M_ + b] = best;                      // best_score
        int tag = bt;
        tagbuf[W_ - 1] = tag;
        for (int t = W_ - 2; t >= 0; --t) {
            int prev = bp[((size_t)b * W_ + (t + 1)) * C_ + tag];
            if (t < seq_end) tag = prev;
            tagbuf[t] = tag;
        }
    }
    __syncthreads();
    for (int t = tid; t < W_; t += blockDim.x) {
        int m = mask[b * W_ + t];
        out[b * W_ + t]      = m ? (float)tagbuf[t] : -1.0f;   // tags
        out[M_ + b * W_ + t] = m ? 1.0f : 0.0f;                // word_mask
    }
}

// ---------- launch ----------
extern "C" void kernel_launch(void* const* d_in, const int* in_sizes, int n_in,
                              void* d_out, int out_size, void* d_ws, size_t ws_size,
                              hipStream_t stream) {
    const float* features = (const float*)d_in[0];
    const int*   word_idx = (const int*)d_in[1];
    const float* proj_w   = (const float*)d_in[2];
    const float* proj_b   = (const float*)d_in[3];
    const float* w_ih_f   = (const float*)d_in[4];
    const float* w_hh_f   = (const float*)d_in[5];
    const float* b_ih_f   = (const float*)d_in[6];
    const float* b_hh_f   = (const float*)d_in[7];
    const float* w_ih_b   = (const float*)d_in[8];
    const float* w_hh_b   = (const float*)d_in[9];
    const float* b_ih_b   = (const float*)d_in[10];
    const float* b_hh_b   = (const float*)d_in[11];
    const float* cls_w    = (const float*)d_in[12];
    const float* cls_b    = (const float*)d_in[13];
    const float* start_t  = (const float*)d_in[14];
    const float* end_t    = (const float*)d_in[15];
    const float* trans    = (const float*)d_in[16];
    float* out = (float*)d_out;

    size_t off = 0;
    auto alloc = [&](size_t bytes) -> void* {
        off = (off + 255) & ~(size_t)255;
        void* p = (char*)d_ws + off;
        off += bytes;
        return p;
    };
    int*  first   = (int*)alloc(B_ * W_ * 4);
    int*  maski   = (int*)alloc(B_ * W_ * 4);
    u16*  pw_bf   = (u16*)alloc((size_t)P_ * D_ * 2);
    u16*  wihf_bf = (u16*)alloc((size_t)G4_ * P_ * 2);
    u16*  wihb_bf = (u16*)alloc((size_t)G4_ * P_ * 2);
    u16*  whhf_bf = (u16*)alloc((size_t)G4_ * H_ * 2);
    u16*  whhb_bf = (u16*)alloc((size_t)G4_ * H_ * 2);
    u16*  x_bf    = (u16*)alloc((size_t)M_ * P_ * 2);
    float* xg_f   = (float*)alloc((size_t)M_ * G4_ * 4);
    float* xg_b   = (float*)alloc((size_t)M_ * G4_ * 4);
    float* hf     = (float*)alloc((size_t)M_ * H_ * 4);
    float* hb     = (float*)alloc((size_t)M_ * H_ * 4);
    float* emis   = (float*)alloc((size_t)M_ * C_ * 4);
    int*   bp     = (int*)alloc((size_t)M_ * C_ * 4);

    // word gather indices + mask
    k_init_first<<<(B_ * W_ + 255) / 256, 256, 0, stream>>>(first);
    k_scan_first<<<(B_ * S_ + 255) / 256, 256, 0, stream>>>(word_idx, first);
    k_mask<<<(B_ * W_ + 255) / 256, 256, 0, stream>>>(first, maski);

    // weight conversions to bf16
    k_f2bf_arr<<<((P_ * D_) + 255) / 256, 256, 0, stream>>>(proj_w, pw_bf, P_ * D_);
    k_f2bf_arr<<<((G4_ * P_) + 255) / 256, 256, 0, stream>>>(w_ih_f, wihf_bf, G4_ * P_);
    k_f2bf_arr<<<((G4_ * P_) + 255) / 256, 256, 0, stream>>>(w_ih_b, wihb_bf, G4_ * P_);
    k_f2bf_arr<<<((G4_ * H_) + 255) / 256, 256, 0, stream>>>(w_hh_f, whhf_bf, G4_ * H_);
    k_f2bf_arr<<<((G4_ * H_) + 255) / 256, 256, 0, stream>>>(w_hh_b, whhb_bf, G4_ * H_);

    // fused gather + projection + GELU (WMMA), out bf16
    k_proj_gemm<<<dim3(P_ / 512, M_ / 32), 256, 0, stream>>>(
        features, first, pw_bf, proj_b, x_bf);

    // LSTM input gates: xg = x @ w_ih^T + b_ih + b_hh (WMMA)
    k_gemm_bf16<P_, G4_><<<dim3(G4_ / 512, M_ / 32), 256, 0, stream>>>(
        x_bf, wihf_bf, b_ih_f, b_hh_f, xg_f);
    k_gemm_bf16<P_, G4_><<<dim3(G4_ / 512, M_ / 32), 256, 0, stream>>>(
        x_bf, wihb_bf, b_ih_b, b_hh_b, xg_b);

    // recurrent BiLSTM (persistent, WMMA per step)
    k_lstm<<<2, 256, 0, stream>>>(xg_f, xg_b, whhf_bf, whhb_bf, hf, hb);

    // classifier + viterbi
    k_cls<<<(M_ + 127) / 128, 128, 0, stream>>>(hf, hb, cls_w, cls_b, emis);
    k_viterbi<<<B_, 32, 0, stream>>>(emis, maski, start_t, end_t, trans, bp, out);
}
